// AttentionBlock_6700148982243
// MI455X (gfx1250) — compile-verified, matrix operand order checked by reading, not run on hardware
//
#include <hip/hip_runtime.h>
#include <hip/hip_bf16.h>
#include <math.h>

// Problem constants (match reference)
#define BB 4
#define SS 2048
#define DMODEL 2048
#define NH 16
#define DH 128

typedef __bf16 bf16_t;
typedef __attribute__((ext_vector_type(16))) __bf16 v16bf;
typedef __attribute__((ext_vector_type(8)))  __bf16 v8bf;
typedef __attribute__((ext_vector_type(8)))  float  v8f;

__device__ __forceinline__ v8f wmma_bf16(v16bf a, v16bf b, v8f c) {
  return __builtin_amdgcn_wmma_f32_16x16x32_bf16(false, a, false, b, (short)0, c,
                                                 false, false);
}

__device__ __forceinline__ v16bf load_frag(const bf16_t* p) {
  v8bf lo = *reinterpret_cast<const v8bf*>(p);
  v8bf hi = *reinterpret_cast<const v8bf*>(p + 8);
  return __builtin_shufflevector(lo, hi, 0, 1, 2, 3, 4, 5, 6, 7,
                                 8, 9, 10, 11, 12, 13, 14, 15);
}

__device__ __forceinline__ v8f zero8() {
  v8f z = {0.f, 0.f, 0.f, 0.f, 0.f, 0.f, 0.f, 0.f};
  return z;
}

// gfx1250 async copy: global -> LDS, 16 bytes per lane, tracked by ASYNCcnt.
__device__ __forceinline__ void async_b128(bf16_t* lds, const bf16_t* g) {
  uint32_t l = (uint32_t)(size_t)lds;
  asm volatile("global_load_async_to_lds_b128 %0, %1, off"
               :: "v"(l), "v"(g) : "memory");
}

template <int N>
__device__ __forceinline__ void wait_async() {
  asm volatile("s_wait_asynccnt %0" :: "i"(N) : "memory");
}

// ---------------------------------------------------------------------------
// Block-tiled 128x128 GEMM mainloop, 4-buffer ring, 2 stages of async copies
// in flight.  A: row-major [128][K] ld=DMODEL.  B: n-major [128][K] ld=DMODEL.
// Each of 8 waves accumulates a 16(M)x128(N) strip into acc[8].
// LDS tile layout [row][32] bf16 == fragment layout.
__device__ __forceinline__ void gemm_mainloop_128(
    const bf16_t* __restrict__ Ab, const bf16_t* __restrict__ Bb,
    bf16_t* Alds, bf16_t* Blds, v8f acc[8]) {
  const int t     = threadIdx.x;
  const int lane  = t & 31;
  const int w     = t >> 5;
  const int col   = lane & 15;
  const int khalf = (lane >> 4) * 16;

  // 512 16-byte chunks per 8KB tile; thread stages chunks t and t+256.
  const int c0 = t, c1 = t + 256;
  const size_t ga0 = (size_t)(c0 >> 2) * DMODEL + (c0 & 3) * 8;
  const size_t ga1 = (size_t)(c1 >> 2) * DMODEL + (c1 & 3) * 8;

  const int NK = DMODEL / 32;  // 64 K-steps

  // Prologue: stages 0 and 1 (8 async instrs/wave in flight).
#pragma unroll
  for (int s = 0; s < 2; ++s) {
    bf16_t* Abuf = Alds + s * (128 * 32);
    bf16_t* Bbuf = Blds + s * (128 * 32);
    const int ko = s * 32;
    async_b128(Abuf + c0 * 8, Ab + ko + ga0);
    async_b128(Abuf + c1 * 8, Ab + ko + ga1);
    async_b128(Bbuf + c0 * 8, Bb + ko + ga0);
    async_b128(Bbuf + c1 * 8, Bb + ko + ga1);
  }

  for (int kk = 0; kk < NK; ++kk) {
    bf16_t* Ac = Alds + (kk & 3) * (128 * 32);
    bf16_t* Bc = Blds + (kk & 3) * (128 * 32);
    if (kk + 2 < NK) {
      bf16_t* An = Alds + ((kk + 2) & 3) * (128 * 32);
      bf16_t* Bn = Blds + ((kk + 2) & 3) * (128 * 32);
      const int ko = (kk + 2) * 32;
      async_b128(An + c0 * 8, Ab + ko + ga0);
      async_b128(An + c1 * 8, Ab + ko + ga1);
      async_b128(Bn + c0 * 8, Bb + ko + ga0);
      async_b128(Bn + c1 * 8, Bb + ko + ga1);
      wait_async<8>();          // stage kk landed; kk+1,kk+2 still in flight
    } else if (kk + 1 < NK) {
      wait_async<4>();
    } else {
      wait_async<0>();
    }
    __syncthreads();            // stage kk visible to all waves

    v16bf a = load_frag(Ac + (w * 16 + col) * 32 + khalf);
    v16bf bfr[8];
#pragma unroll
    for (int nt = 0; nt < 8; ++nt)
      bfr[nt] = load_frag(Bc + (nt * 16 + col) * 32 + khalf);
#pragma unroll
    for (int nt = 0; nt < 8; ++nt)
      acc[nt] = wmma_bf16(a, bfr[nt], acc[nt]);

    __syncthreads();            // readers done before ring slot is rewritten
  }
}

// ---------------------------------------------------------------------------
// Kernel 1a: fp32 -> bf16 elementwise convert
__global__ void k_cvt(const float* __restrict__ src, bf16_t* __restrict__ dst, int n) {
  int i = blockIdx.x * blockDim.x + threadIdx.x;
  if (i < n) dst[i] = (bf16_t)src[i];
}

// Kernel 1b: transpose-convert weights: dst[h][n][k] = (bf16) src[h][k][n]
__global__ void k_tr(const float* __restrict__ src, bf16_t* __restrict__ dst,
                     int H_, int K_, int N_) {
  int i = blockIdx.x * blockDim.x + threadIdx.x;
  int total = H_ * K_ * N_;
  if (i >= total) return;
  int h = i / (K_ * N_);
  int r = i - h * (K_ * N_);
  int n = r / K_;
  int kx = r - n * K_;
  dst[i] = (bf16_t)src[(size_t)h * K_ * N_ + (size_t)kx * N_ + n];
}

// ---------------------------------------------------------------------------
// Kernel 2: QKV projection.  One block: 128(M rows of x) x 128(= DH) for one
// of {q,k,v} and one (b,h).  q,k stored [B,S,H,DH]; v stored transposed
// [B,H,DH,S] so the PV B-operand is contiguous later.
__global__ void __launch_bounds__(256)
k_qkv(const bf16_t* __restrict__ xb, const bf16_t* __restrict__ Wt,
      const float* __restrict__ Qb, const float* __restrict__ Kb,
      const float* __restrict__ Vb,
      bf16_t* __restrict__ qout, bf16_t* __restrict__ kout,
      bf16_t* __restrict__ vT) {
  __shared__ bf16_t Alds[4 * 128 * 32];
  __shared__ bf16_t Blds[4 * 128 * 32];

  const int MTB = SS / 128;               // 16 M-blocks per (b,h)
  const int per = BB * NH * MTB;          // 1024 blocks per tensor
  const int bid = blockIdx.x;
  const int which = bid / per;            // 0=q 1=k 2=v
  int r = bid - which * per;
  const int b = r / (NH * MTB);
  r -= b * (NH * MTB);
  const int h  = r / MTB;
  const int m0 = (r - h * MTB) * 128;

  const int lane   = threadIdx.x & 31;
  const int w      = threadIdx.x >> 5;
  const int col    = lane & 15;
  const int rowoff = (lane >> 4) * 8;

  const bf16_t* Ab = xb + (size_t)(b * SS + m0) * DMODEL;
  const bf16_t* Bb = Wt + (size_t)which * NH * DH * DMODEL
                        + (size_t)h * DH * DMODEL;

  v8f acc[8];
#pragma unroll
  for (int t = 0; t < 8; ++t) acc[t] = zero8();

  gemm_mainloop_128(Ab, Bb, Alds, Blds, acc);

  const float* bias = (which == 0) ? Qb : (which == 1) ? Kb : Vb;
#pragma unroll
  for (int nt = 0; nt < 8; ++nt) {
    float bv = bias[h * DH + nt * 16 + col];
#pragma unroll
    for (int i = 0; i < 8; ++i) {
      float o = acc[nt][i] + bv;
      int row = m0 + w * 16 + i + rowoff;
      if (which == 2) {
        vT[(size_t)((b * NH + h) * DH + nt * 16 + col) * SS + row] = (bf16_t)o;
      } else {
        bf16_t* dst = (which == 0) ? qout : kout;
        dst[(size_t)((b * SS + row) * NH + h) * DH + nt * 16 + col] = (bf16_t)o;
      }
    }
  }
}

// ---------------------------------------------------------------------------
// Kernel 3: block-cooperative causal flash attention.
// One block = 8 consecutive 16-row q tiles (128 q rows) of one (b,h); all 8
// waves share double-buffered async-staged K and V^T tiles (32 keys each).
// Trip count is block-uniform; fully-masked blocks are numerically inert.
__global__ void __launch_bounds__(256)
k_flash(const bf16_t* __restrict__ q, const bf16_t* __restrict__ k,
        const bf16_t* __restrict__ vT, bf16_t* __restrict__ attn) {
  __shared__ bf16_t Kt[2][32 * 128];      // [key][dim]   8KB x2
  __shared__ bf16_t Vt[2][128 * 32];      // [dim][key]   8KB x2
  __shared__ bf16_t plds[8][16 * 32];     // per-wave P staging, 8KB

  const int bid = blockIdx.x;             // B*H*(S/128) = 1024
  const int bq = bid & 15;                // q-block within sequence
  const int h  = (bid >> 4) & 15;
  const int b  = bid >> 8;

  const int t      = threadIdx.x;
  const int w      = t >> 5;
  const int lane   = t & 31;
  const int col    = lane & 15;
  const int khalf  = (lane >> 4) * 16;
  const int rowoff = (lane >> 4) * 8;
  const int q0     = (bq * 8 + w) * 16;
  const float scale = 0.08838834764831845f;  // 1/sqrt(128)

  const bf16_t* Kg = k  + ((size_t)b * SS * NH + h) * DH;   // + key*NH*DH + dim
  const bf16_t* Vg = vT + ((size_t)(b * NH + h)) * DH * SS; // + dim*SS + key

  // staging: K tile 512 chunks (key=ch>>4, off=(ch&15)*8);
  //          V tile 512 chunks (dim=ch>>2, off=(ch&3)*8); LDS offset = ch*8.
  const int c0 = t, c1 = t + 256;

  v16bf Qf[4];
#pragma unroll
  for (int c = 0; c < 4; ++c)
    Qf[c] = load_frag(q + (size_t)((b * SS + q0 + col) * NH + h) * DH + c * 32 + khalf);

  v8f Oacc[8];
  float mrun[8], lrun[8];
#pragma unroll
  for (int i = 0; i < 8; ++i) { Oacc[i] = zero8(); mrun[i] = -INFINITY; lrun[i] = 0.f; }

  const int nkb = 4 * bq + 4;             // covers keys <= last row of block

  // Prologue: stage key-block 0 into buffer 0.
  async_b128(&Kt[0][c0 * 8], Kg + (size_t)(c0 >> 4) * (NH * DH) + (c0 & 15) * 8);
  async_b128(&Kt[0][c1 * 8], Kg + (size_t)(c1 >> 4) * (NH * DH) + (c1 & 15) * 8);
  async_b128(&Vt[0][c0 * 8], Vg + (size_t)(c0 >> 2) * SS + (c0 & 3) * 8);
  async_b128(&Vt[0][c1 * 8], Vg + (size_t)(c1 >> 2) * SS + (c1 & 3) * 8);

  for (int jb = 0; jb < nkb; ++jb) {
    const int kb  = jb * 32;
    const int cur = jb & 1;
    if (jb + 1 < nkb) {
      const int nxt = (jb + 1) & 1;
      const int kn  = kb + 32;
      async_b128(&Kt[nxt][c0 * 8], Kg + (size_t)(kn + (c0 >> 4)) * (NH * DH) + (c0 & 15) * 8);
      async_b128(&Kt[nxt][c1 * 8], Kg + (size_t)(kn + (c1 >> 4)) * (NH * DH) + (c1 & 15) * 8);
      async_b128(&Vt[nxt][c0 * 8], Vg + (size_t)(c0 >> 2) * SS + kn + (c0 & 3) * 8);
      async_b128(&Vt[nxt][c1 * 8], Vg + (size_t)(c1 >> 2) * SS + kn + (c1 & 3) * 8);
      wait_async<4>();
    } else {
      wait_async<0>();
    }
    __syncthreads();

    // S = Q x K^T for 32 keys: load all 8 K fragments, then 8 WMMAs.
    v16bf kf[8];
#pragma unroll
    for (int c = 0; c < 4; ++c) {
      kf[c]     = load_frag(&Kt[cur][(col)      * 128 + c * 32 + khalf]);
      kf[4 + c] = load_frag(&Kt[cur][(col + 16) * 128 + c * 32 + khalf]);
    }
    v8f s0 = zero8(), s1 = zero8();
#pragma unroll
    for (int c = 0; c < 4; ++c) {
      s0 = wmma_bf16(Qf[c], kf[c],     s0);
      s1 = wmma_bf16(Qf[c], kf[4 + c], s1);
    }

    // online softmax; rows live in VGPR index i (+8 for upper half-wave)
#pragma unroll
    for (int i = 0; i < 8; ++i) {
      const int qpos = q0 + i + rowoff;
      float x0 = s0[i] * scale; if (kb + col > qpos)      x0 = -100000.0f;
      float x1 = s1[i] * scale; if (kb + 16 + col > qpos) x1 = -100000.0f;
      float mx = fmaxf(x0, x1);
      for (int d = 1; d < 16; d <<= 1) mx = fmaxf(mx, __shfl_xor(mx, d, 32));
      float mnew = fmaxf(mrun[i], mx);
      float corr = __expf(mrun[i] - mnew);
      float p0 = __expf(x0 - mnew);
      float p1 = __expf(x1 - mnew);
      float rs = p0 + p1;
      for (int d = 1; d < 16; d <<= 1) rs += __shfl_xor(rs, d, 32);
      lrun[i] = lrun[i] * corr + rs;
      mrun[i] = mnew;
#pragma unroll
      for (int nt = 0; nt < 8; ++nt) Oacc[nt][i] *= corr;
      plds[w][(i + rowoff) * 32 + col]      = (bf16_t)p0;
      plds[w][(i + rowoff) * 32 + 16 + col] = (bf16_t)p1;
    }
    asm volatile("s_wait_dscnt 0" ::: "memory");   // same-wave LDS RAW
    v16bf Pf = load_frag(&plds[w][col * 32 + khalf]);  // C-layout -> A-layout

    // O += P x V: load all 8 V fragments, then 8 WMMAs.
    v16bf vf[8];
#pragma unroll
    for (int nt = 0; nt < 8; ++nt)
      vf[nt] = load_frag(&Vt[cur][(nt * 16 + col) * 32 + khalf]);
#pragma unroll
    for (int nt = 0; nt < 8; ++nt)
      Oacc[nt] = wmma_bf16(Pf, vf[nt], Oacc[nt]);

    __syncthreads();            // readers done before buffer is rewritten
  }

#pragma unroll
  for (int i = 0; i < 8; ++i) {
    float inv = 1.0f / lrun[i];
    int row = q0 + i + rowoff;
#pragma unroll
    for (int nt = 0; nt < 8; ++nt) {
      attn[(size_t)(b * SS + row) * DMODEL + h * DH + nt * 16 + col] =
          (bf16_t)(Oacc[nt][i] * inv);
    }
  }
}

// ---------------------------------------------------------------------------
// Kernel 4: output projection.  One block: 128x128 tile of out.
// attn[B*S, 2048] x OwT[n][k] + O_b -> fp32 out.
__global__ void __launch_bounds__(256)
k_oproj(const bf16_t* __restrict__ attn, const bf16_t* __restrict__ OwT,
        const float* __restrict__ Ob, float* __restrict__ out) {
  __shared__ bf16_t Alds[4 * 128 * 32];
  __shared__ bf16_t Blds[4 * 128 * 32];

  const int bid = blockIdx.x;
  const int ns  = bid & 15;               // 16 column blocks of 128
  const int m0  = (bid >> 4) * 128;       // 64 row blocks

  const int lane   = threadIdx.x & 31;
  const int w      = threadIdx.x >> 5;
  const int col    = lane & 15;
  const int rowoff = (lane >> 4) * 8;

  const bf16_t* Ab = attn + (size_t)m0 * DMODEL;
  const bf16_t* Bb = OwT + (size_t)(ns * 128) * DMODEL;

  v8f acc[8];
#pragma unroll
  for (int t = 0; t < 8; ++t) acc[t] = zero8();

  gemm_mainloop_128(Ab, Bb, Alds, Blds, acc);

#pragma unroll
  for (int nt = 0; nt < 8; ++nt) {
    int n = ns * 128 + nt * 16 + col;
    float bv = Ob[n];
#pragma unroll
    for (int i = 0; i < 8; ++i) {
      int row = m0 + w * 16 + i + rowoff;
      out[(size_t)row * DMODEL + n] = acc[nt][i] + bv;
    }
  }
}

// ---------------------------------------------------------------------------
extern "C" void kernel_launch(void* const* d_in, const int* in_sizes, int n_in,
                              void* d_out, int out_size, void* d_ws, size_t ws_size,
                              hipStream_t stream) {
  const float* x  = (const float*)d_in[0];
  const float* Qw = (const float*)d_in[1];
  const float* Qb = (const float*)d_in[2];
  const float* Kw = (const float*)d_in[3];
  const float* Kb = (const float*)d_in[4];
  const float* Vw = (const float*)d_in[5];
  const float* Vb = (const float*)d_in[6];
  const float* Ow = (const float*)d_in[7];
  const float* Ob = (const float*)d_in[8];
  float* out = (float*)d_out;

  // workspace carve-up (bf16 tensors), ~192 MiB total
  char* p = (char*)d_ws;
  bf16_t* xb  = (bf16_t*)p; p += (size_t)BB * SS * DMODEL * 2;        // 32 MiB
  bf16_t* Wt  = (bf16_t*)p; p += (size_t)3 * NH * DMODEL * DH * 2;    // 24 MiB (q,k,v)
  bf16_t* OwT = (bf16_t*)p; p += (size_t)DMODEL * NH * DH * 2;        //  8 MiB
  bf16_t* qb  = (bf16_t*)p; p += (size_t)BB * SS * NH * DH * 2;       // 32 MiB
  bf16_t* kb  = (bf16_t*)p; p += (size_t)BB * SS * NH * DH * 2;       // 32 MiB
  bf16_t* vT  = (bf16_t*)p; p += (size_t)BB * NH * DH * SS * 2;       // 32 MiB
  bf16_t* at  = (bf16_t*)p; p += (size_t)BB * SS * NH * DH * 2;       // 32 MiB
  (void)ws_size; (void)in_sizes; (void)n_in; (void)out_size;

  const int nX = BB * SS * DMODEL;
  k_cvt<<<(nX + 255) / 256, 256, 0, stream>>>(x, xb, nX);

  const int nW = NH * DMODEL * DH;
  k_tr<<<(nW + 255) / 256, 256, 0, stream>>>(Qw, Wt + 0 * (size_t)nW, NH, DMODEL, DH);
  k_tr<<<(nW + 255) / 256, 256, 0, stream>>>(Kw, Wt + 1 * (size_t)nW, NH, DMODEL, DH);
  k_tr<<<(nW + 255) / 256, 256, 0, stream>>>(Vw, Wt + 2 * (size_t)nW, NH, DMODEL, DH);
  const int nO = DMODEL * NH * DH;
  k_tr<<<(nO + 255) / 256, 256, 0, stream>>>(Ow, OwT, 1, NH * DH, DMODEL);

  // 3 tensors * B*H*(S/128) = 3072 blocks, each a 128x128 output tile
  k_qkv<<<3 * BB * NH * (SS / 128), 256, 0, stream>>>(xb, Wt, Qb, Kb, Vb,
                                                      qb, kb, vT);
  // B*H*(S/128) = 1024 blocks, 8 q-tiles each
  k_flash<<<BB * NH * (SS / 128), 256, 0, stream>>>(qb, kb, vT, at);
  // (B*S/128)*(DMODEL/128) = 1024 blocks
  k_oproj<<<(BB * SS / 128) * (DMODEL / 128), 256, 0, stream>>>(at, OwT, Ob, out);
}
